// Attention_14113262534656
// MI455X (gfx1250) — matrix operand order repaired and checked by reading, not compiled, not run on hardware
//
#include <hip/hip_runtime.h>
#include <hip/hip_bf16.h>

// ---------------------------------------------------------------------------
// MHA forward for MI455X (gfx1250, wave32, WMMA f32_16x16x32_f16)
// B=4, S=2048, D_MODEL=1024, H=16, d_head=64
// 128-bit fragment traffic; async global->LDS DMA for K/V staging;
// GEMMs: block tile 128x128, wave tile 32x64 (8 WMMAs / k-step / wave).
// ---------------------------------------------------------------------------

#define SEQ    2048
#define BATCH  4
#define NHEAD  16
#define DHEAD  64
#define DMODEL 1024
#define ROWS   (BATCH * SEQ)   // 8192

typedef __attribute__((ext_vector_type(16))) _Float16 v16h;
typedef __attribute__((ext_vector_type(8)))  float    v8f;

union Frag {
    v16h v;
    unsigned int u[8];
    uint4 q[2];
};

// A-matrix (16-bit, 16x32): lane group g: halves 0..7 are K=(0..7)+8g,
// halves 8..15 are K=(16..23)+8g -> two contiguous 16B runs at halves 8g and 16+8g.
__device__ __forceinline__ void load_afrag(Frag& f, const _Float16* rowbase, int grp) {
    f.q[0] = *(const uint4*)(rowbase + 8 * grp);
    f.q[1] = *(const uint4*)(rowbase + 16 + 8 * grp);
}
// B-matrix (16-bit, 32x16): same K order as A: halves 0..7 are K=(0..7)+8g,
// halves 8..15 are K=(16..23)+8g -> two 16B runs at halves 8g and 16+8g.
__device__ __forceinline__ void load_bfrag(Frag& f, const _Float16* rowbase, int grp) {
    f.q[0] = *(const uint4*)(rowbase + 8 * grp);
    f.q[1] = *(const uint4*)(rowbase + 16 + 8 * grp);
}

// 16B async DMA global -> LDS (GLOBAL_LOAD_ASYNC_TO_LDS_B128, ASYNCcnt-tracked)
__device__ __forceinline__ void async_copy16(const void* g, void* lds) {
    unsigned loff = (unsigned)(size_t)lds;   // generic LDS ptr: low 32 bits = LDS offset
    asm volatile("global_load_async_to_lds_b128 %0, %1, off"
                 :: "v"(loff), "v"(g) : "memory");
}
__device__ __forceinline__ void wait_async0() {
#if __has_builtin(__builtin_amdgcn_s_wait_asynccnt)
    __builtin_amdgcn_s_wait_asynccnt(0);
#else
    asm volatile("s_wait_asynccnt 0" ::: "memory");
#endif
}

// ---------------------------------------------------------------------------
// Kernel 0: W [in][out] f32 -> Wt [out][in] f16 (coalesced dword writes)
// ---------------------------------------------------------------------------
__global__ __launch_bounds__(256) void wtrans_kernel(const float* __restrict__ W,
                                                     _Float16* __restrict__ Wt) {
    int idx = blockIdx.x * 256 + threadIdx.x;       // 512K pair-tasks
    int o = idx >> 9, kp = idx & 511;
    union { _Float16 h[2]; unsigned int u; } t;
    t.h[0] = (_Float16)W[(size_t)(2 * kp) * DMODEL + o];
    t.h[1] = (_Float16)W[(size_t)(2 * kp + 1) * DMODEL + o];
    *(unsigned int*)(Wt + (size_t)o * DMODEL + 2 * kp) = t.u;
}

// ---------------------------------------------------------------------------
// Kernel 1: Y = X(f32) @ Wt^T + bias -> f16, head-split layout
//   vmode 0: out[b][h][m][d] (Q,K)   vmode 1: out[b][h][d][m] (V transposed)
// Block 256 thr = 8 waves; tile M=128,N=128,Kstep=32; wave = 32x64 (2x4 WMMA)
// ---------------------------------------------------------------------------
#define LDA 40   // halves; 80B row stride (16B multiple)
__global__ __launch_bounds__(256) void proj_gemm_f32(const float* __restrict__ X,
                                                     const _Float16* __restrict__ Wt,
                                                     const float* __restrict__ bias,
                                                     _Float16* __restrict__ out,
                                                     int vmode) {
    __shared__ _Float16 As[128 * LDA];
    __shared__ _Float16 Bs[128 * LDA];
    const int tid = threadIdx.x;
    const int lane = tid & 31, wave = tid >> 5;
    const int grp = lane >> 4, ln16 = lane & 15;
    const int wm = wave & 3, wn = wave >> 2;        // 4 waves x 32 rows, 2 waves x 64 cols
    const int m0 = blockIdx.y * 128;
    const int n0 = blockIdx.x * 128;

    v8f c[2][4] = {};
    for (int k0 = 0; k0 < DMODEL; k0 += 32) {
        __syncthreads();
        // stage X tile 128x32 f32 as float4, convert -> f16 (uint2 LDS stores)
        for (int i = 0; i < 4; ++i) {
            int e = i * 256 + tid;                  // 1024 float4 chunks
            int r = e >> 3, q4 = e & 7;
            float4 f = *(const float4*)(X + (size_t)(m0 + r) * DMODEL + k0 + q4 * 4);
            union { _Float16 h[4]; uint2 u2; } t;
            t.h[0] = (_Float16)f.x; t.h[1] = (_Float16)f.y;
            t.h[2] = (_Float16)f.z; t.h[3] = (_Float16)f.w;
            *(uint2*)&As[r * LDA + q4 * 4] = t.u2;
        }
        // stage Wt tile 128x32 halves as b128 (512 chunks, 2/thread)
        for (int i = 0; i < 2; ++i) {
            int e = i * 256 + tid;
            int r = e >> 2, ch = e & 3;
            uint4 val = *(const uint4*)(Wt + (size_t)(n0 + r) * DMODEL + k0 + ch * 8);
            *(uint4*)&Bs[r * LDA + ch * 8] = val;
        }
        __syncthreads();
        Frag a[2], b[4];
        for (int mi = 0; mi < 2; ++mi)
            load_afrag(a[mi], &As[(wm * 32 + mi * 16 + ln16) * LDA], grp);
        for (int ni = 0; ni < 4; ++ni)
            load_bfrag(b[ni], &Bs[(wn * 64 + ni * 16 + ln16) * LDA], grp);
        for (int mi = 0; mi < 2; ++mi)
            for (int ni = 0; ni < 4; ++ni)
                c[mi][ni] = __builtin_amdgcn_wmma_f32_16x16x32_f16(
                    false, a[mi].v, false, b[ni].v, (short)0, c[mi][ni], false, false);
    }
    for (int mi = 0; mi < 2; ++mi)
        for (int ni = 0; ni < 4; ++ni)
            for (int j = 0; j < 8; ++j) {
                int row = m0 + wm * 32 + mi * 16 + j + 8 * grp;
                int col = n0 + wn * 64 + ni * 16 + ln16;
                float val = c[mi][ni][j] + bias[col];
                int b_ = row >> 11, m = row & (SEQ - 1);
                int h = col >> 6, dd = col & (DHEAD - 1);
                size_t idx;
                if (vmode == 0)
                    idx = (((size_t)(b_ * NHEAD + h)) * SEQ + m) * DHEAD + dd;
                else
                    idx = (((size_t)(b_ * NHEAD + h)) * DHEAD + dd) * SEQ + m;
                out[idx] = (_Float16)val;
            }
}

// ---------------------------------------------------------------------------
// Kernel 2: causal flash attention, softmax((QK^T - 1e10*triu)/8).
// Block 256 thr (8 waves) = 128 query rows of one (b,h); per 32-key block:
// async-DMA K/V into LDS, 4 WMMAs QK^T, online softmax, 4 WMMAs P.V.
// ---------------------------------------------------------------------------
#define LDK 72   // Ks row stride (halves): 144B, 16B multiple
#define LDV 40   // Vs/Pb row stride (halves): 80B, 16B multiple
__global__ __launch_bounds__(256) void flash_attn_kernel(const _Float16* __restrict__ Qh,
                                                         const _Float16* __restrict__ Kh,
                                                         const _Float16* __restrict__ Vt,
                                                         _Float16* __restrict__ O) {
    __shared__ _Float16 Ks[32 * LDK];            // 32 keys x 64 d
    __shared__ _Float16 Vs[64 * LDV];            // 64 d x 32 keys (V pre-transposed)
    __shared__ _Float16 Pb[8 * 16 * LDV];        // per-wave P re-layout buffer

    const int tid = threadIdx.x;
    const int lane = tid & 31, wave = tid >> 5;
    const int grp = lane >> 4, ln16 = lane & 15;
    const int bh = blockIdx.y;
    const int m0 = blockIdx.x * 128;

    // Q fragments: rows m0+wave*16.., K-dim d=64 -> 2 fragments, b128 loads
    const _Float16* qbase = Qh + ((size_t)bh * SEQ + m0 + wave * 16 + ln16) * DHEAD;
    Frag aq[2];
    load_afrag(aq[0], qbase, grp);
    load_afrag(aq[1], qbase + 32, grp);

    v8f acc[4] = {};
    float mi[8], li[8];
    for (int j = 0; j < 8; ++j) { mi[j] = -3.0e38f; li[j] = 0.0f; }

    // per-thread DMA chunk assignment (16B each, whole block covers both tiles)
    const int kr = tid >> 3, kc = tid & 7;       // Ks: 32 rows x 8 chunks
    const int vr = tid >> 2, vc = tid & 3;       // Vs: 64 rows x 4 chunks
    const _Float16* ksrc = Kh + ((size_t)bh * SEQ + kr) * DHEAD + kc * 8;
    const _Float16* vsrc = Vt + ((size_t)bh * DHEAD + vr) * SEQ + vc * 8;
    _Float16* kdst = &Ks[kr * LDK + kc * 8];
    _Float16* vdst = &Vs[vr * LDV + vc * 8];

    const int nend = m0 + 128;                   // causal upper bound
    for (int n0 = 0; n0 < nend; n0 += 32) {
        __syncthreads();                         // all readers of Ks/Vs done
        async_copy16(ksrc + (size_t)n0 * DHEAD, kdst);
        async_copy16(vsrc + n0, vdst);
        if (n0 + 32 < nend) {                    // prefetch next block
            __builtin_prefetch(ksrc + (size_t)(n0 + 32) * DHEAD, 0, 0);
            __builtin_prefetch(vsrc + n0 + 32, 0, 0);
        }
        wait_async0();
        __syncthreads();

        // scores: two 16x16 tiles over d=64 (2 WMMAs each)
        v8f s[2] = {};
        for (int t = 0; t < 2; ++t) {
            Frag bk0, bk1;
            const _Float16* krow = &Ks[(t * 16 + ln16) * LDK];
            load_bfrag(bk0, krow, grp);
            load_bfrag(bk1, krow + 32, grp);
            s[t] = __builtin_amdgcn_wmma_f32_16x16x32_f16(false, aq[0].v, false, bk0.v,
                                                          (short)0, s[t], false, false);
            s[t] = __builtin_amdgcn_wmma_f32_16x16x32_f16(false, aq[1].v, false, bk1.v,
                                                          (short)0, s[t], false, false);
        }

        // causal mask (before scale, as reference) + online softmax
        float corr[8];
        for (int j = 0; j < 8; ++j) {
            int row = m0 + wave * 16 + j + 8 * grp;
            float x0 = s[0][j];
            float x1 = s[1][j];
            if (n0 + ln16 > row)      x0 -= 1.0e10f;
            if (n0 + 16 + ln16 > row) x1 -= 1.0e10f;
            x0 *= 0.125f;
            x1 *= 0.125f;
            float rm = fmaxf(x0, x1);
            for (int msk = 8; msk >= 1; msk >>= 1)
                rm = fmaxf(rm, __shfl_xor(rm, msk, 32));
            float nm = fmaxf(mi[j], rm);
            float cr = __expf(mi[j] - nm);
            float p0 = __expf(x0 - nm);
            float p1 = __expf(x1 - nm);
            float rs = p0 + p1;
            for (int msk = 8; msk >= 1; msk >>= 1)
                rs += __shfl_xor(rs, msk, 32);
            li[j] = li[j] * cr + rs;
            mi[j] = nm;
            corr[j] = cr;
            s[0][j] = p0;
            s[1][j] = p1;
        }
        for (int nt = 0; nt < 4; ++nt)
            for (int j = 0; j < 8; ++j) acc[nt][j] *= corr[j];

        // re-layout P (C layout -> A fragment) via per-wave LDS buffer
        _Float16* pb = Pb + wave * 16 * LDV;
        for (int j = 0; j < 8; ++j) {
            pb[(j + 8 * grp) * LDV + ln16]      = (_Float16)s[0][j];
            pb[(j + 8 * grp) * LDV + 16 + ln16] = (_Float16)s[1][j];
        }
        __syncthreads();

        Frag ap;
        load_afrag(ap, &pb[ln16 * LDV], grp);
        for (int nt = 0; nt < 4; ++nt) {
            Frag bv;
            load_bfrag(bv, &Vs[(nt * 16 + ln16) * LDV], grp);
            acc[nt] = __builtin_amdgcn_wmma_f32_16x16x32_f16(false, ap.v, false, bv.v,
                                                             (short)0, acc[nt], false, false);
        }
    }

    // normalize, store O f16 [B][S][H*64]
    size_t b_ = bh >> 4;
    int h = bh & (NHEAD - 1);
    for (int nt = 0; nt < 4; ++nt)
        for (int j = 0; j < 8; ++j) {
            int m = m0 + wave * 16 + j + 8 * grp;
            int dd = h * DHEAD + nt * 16 + ln16;
            float val = acc[nt][j] / li[j];
            O[((size_t)b_ * SEQ + m) * DMODEL + dd] = (_Float16)val;
        }
}

// ---------------------------------------------------------------------------
// Kernel 3: out = O16(f16) @ WOt^T + bias -> f32 d_out [8192][1024]
// Same 128x128 block / 32x64 wave tiling as Kernel 1.
// ---------------------------------------------------------------------------
__global__ __launch_bounds__(256) void out_gemm_f16(const _Float16* __restrict__ A,
                                                    const _Float16* __restrict__ Wt,
                                                    const float* __restrict__ bias,
                                                    float* __restrict__ out) {
    __shared__ _Float16 As[128 * LDA];
    __shared__ _Float16 Bs[128 * LDA];
    const int tid = threadIdx.x;
    const int lane = tid & 31, wave = tid >> 5;
    const int grp = lane >> 4, ln16 = lane & 15;
    const int wm = wave & 3, wn = wave >> 2;
    const int m0 = blockIdx.y * 128;
    const int n0 = blockIdx.x * 128;

    v8f c[2][4] = {};
    for (int k0 = 0; k0 < DMODEL; k0 += 32) {
        __syncthreads();
        for (int i = 0; i < 2; ++i) {           // A tile: 512 x 16B chunks
            int e = i * 256 + tid;
            int r = e >> 2, ch = e & 3;
            uint4 val = *(const uint4*)(A + (size_t)(m0 + r) * DMODEL + k0 + ch * 8);
            *(uint4*)&As[r * LDA + ch * 8] = val;
        }
        for (int i = 0; i < 2; ++i) {           // B tile: 512 x 16B chunks
            int e = i * 256 + tid;
            int r = e >> 2, ch = e & 3;
            uint4 val = *(const uint4*)(Wt + (size_t)(n0 + r) * DMODEL + k0 + ch * 8);
            *(uint4*)&Bs[r * LDA + ch * 8] = val;
        }
        __syncthreads();
        Frag a[2], b[4];
        for (int mi = 0; mi < 2; ++mi)
            load_afrag(a[mi], &As[(wm * 32 + mi * 16 + ln16) * LDA], grp);
        for (int ni = 0; ni < 4; ++ni)
            load_bfrag(b[ni], &Bs[(wn * 64 + ni * 16 + ln16) * LDA], grp);
        for (int mi = 0; mi < 2; ++mi)
            for (int ni = 0; ni < 4; ++ni)
                c[mi][ni] = __builtin_amdgcn_wmma_f32_16x16x32_f16(
                    false, a[mi].v, false, b[ni].v, (short)0, c[mi][ni], false, false);
    }
    for (int mi = 0; mi < 2; ++mi)
        for (int ni = 0; ni < 4; ++ni)
            for (int j = 0; j < 8; ++j) {
                int row = m0 + wm * 32 + mi * 16 + j + 8 * grp;
                int col = n0 + wn * 64 + ni * 16 + ln16;
                out[(size_t)row * DMODEL + col] = c[mi][ni][j] + bias[col];
            }
}

// ---------------------------------------------------------------------------
// Host-side launch. Workspace: 4x2MB Wt + 16MB Qh + 16MB Kh + 16MB Vt + 16MB O16
// ---------------------------------------------------------------------------
extern "C" void kernel_launch(void* const* d_in, const int* in_sizes, int n_in,
                              void* d_out, int out_size, void* d_ws, size_t ws_size,
                              hipStream_t stream) {
    (void)in_sizes; (void)n_in; (void)out_size; (void)ws_size;
    const float* q   = (const float*)d_in[0];
    const float* k   = (const float*)d_in[1];
    const float* v   = (const float*)d_in[2];
    const float* WQ  = (const float*)d_in[3];
    const float* WQb = (const float*)d_in[4];
    const float* WK  = (const float*)d_in[5];
    const float* WKb = (const float*)d_in[6];
    const float* WV  = (const float*)d_in[7];
    const float* WVb = (const float*)d_in[8];
    const float* WO  = (const float*)d_in[9];
    const float* WOb = (const float*)d_in[10];

    char* ws = (char*)d_ws;
    const size_t MB = 1024 * 1024;
    _Float16* WQt = (_Float16*)(ws + 0 * MB);
    _Float16* WKt = (_Float16*)(ws + 2 * MB);
    _Float16* WVt = (_Float16*)(ws + 4 * MB);
    _Float16* WOt = (_Float16*)(ws + 6 * MB);
    _Float16* Qh  = (_Float16*)(ws + 8 * MB);
    _Float16* Kh  = (_Float16*)(ws + 24 * MB);
    _Float16* Vt  = (_Float16*)(ws + 40 * MB);
    _Float16* O16 = (_Float16*)(ws + 56 * MB);

    wtrans_kernel<<<2048, 256, 0, stream>>>(WQ, WQt);
    wtrans_kernel<<<2048, 256, 0, stream>>>(WK, WKt);
    wtrans_kernel<<<2048, 256, 0, stream>>>(WV, WVt);
    wtrans_kernel<<<2048, 256, 0, stream>>>(WO, WOt);

    dim3 gg(DMODEL / 128, ROWS / 128);  // (8, 64)
    proj_gemm_f32<<<gg, 256, 0, stream>>>(q, WQt, WQb, Qh, 0);
    proj_gemm_f32<<<gg, 256, 0, stream>>>(k, WKt, WKb, Kh, 0);
    proj_gemm_f32<<<gg, 256, 0, stream>>>(v, WVt, WVb, Vt, 1);

    dim3 ga(SEQ / 128, BATCH * NHEAD);  // (16, 64)
    flash_attn_kernel<<<ga, 256, 0, stream>>>(Qh, Kh, Vt, O16);

    out_gemm_f16<<<gg, 256, 0, stream>>>(O16, WOt, WOb, (float*)d_out);
}
